// EmbeddingDecoder_35751307772069
// MI455X (gfx1250) — matrix operand order repaired and checked by reading, not compiled, not run on hardware
//
#include <hip/hip_runtime.h>
#include <stdint.h>

// ---------------------------------------------------------------------------
// 2-layer LSTM decoder (teacher forced) for MI455X / gfx1250, wave32.
// bf16 WMMA (v_wmma_f32_16x16x32_bf16) gate GEMMs, LDS-staged activations,
// 4x M-register-blocking, explicit global-AS loads (no FLAT in hot loop),
// branch-free transcendental epilogue, fp32 cell state.
// ---------------------------------------------------------------------------

#define B_ 256
#define T_ 64
#define H_ 1024
#define Z_ 512
#define E_ 32
#define G_ 4096   // 4*H
#define KC_ 256   // K chunk staged in LDS
#define MT_ 4     // M tiles (16 rows each) per wave

typedef __attribute__((ext_vector_type(16))) __bf16       v16bf;
typedef __attribute__((ext_vector_type(8)))  float        v8f;
typedef __attribute__((ext_vector_type(4)))  unsigned int v4u;   // POD 16B chunk

union AFrag { v4u u[2]; v16bf v; };

// ---- explicit global-address-space accessors (avoid FLAT once LDS exists) ----
#define GAS __attribute__((address_space(1)))
__device__ __forceinline__ v4u ldg_u4(const void* p) {
    return *(const GAS v4u*)(uintptr_t)p;
}
__device__ __forceinline__ float ldg_f(const float* p) {
    return *(const GAS float*)(uintptr_t)p;
}
__device__ __forceinline__ void stg_f(float* p, float v) {
    *(GAS float*)(uintptr_t)p = v;
}
__device__ __forceinline__ void stg_bf(__bf16* p, __bf16 v) {
    *(GAS __bf16*)(uintptr_t)p = v;
}

// ---- branch-free gate nonlinearities (v_exp_f32 + v_rcp_f32, saturating) ----
__device__ __forceinline__ float sigmoid_fast(float x) {
    return __builtin_amdgcn_rcpf(1.0f + __expf(-x));
}
__device__ __forceinline__ float tanh_fast(float x) {
    return 1.0f - 2.0f * __builtin_amdgcn_rcpf(__expf(2.0f * x) + 1.0f);
}

// ---------------------------------------------------------------------------
// Fused LSTM gate GEMM + cell update.
// gates[m, 4H] = A1[m,:K1] @ W1[4H,K1]^T + A2[m,:K2] @ W2[4H,K2]^T + bias
// Wave tile: 64 rows (4 x 16) x 16 hidden units x 4 gates = 16 accumulators.
// Workgroup: 8 waves (same 64 rows, 8 adjacent N tiles); A staged in LDS.
// grid = (8, 4), block = 256. EXEC all-ones at every WMMA.
// ---------------------------------------------------------------------------
__global__ __launch_bounds__(256)
void lstm_gate_wmma(const __bf16* __restrict__ A1, int K1, const __bf16* __restrict__ W1,
                    const __bf16* __restrict__ A2, int K2, const __bf16* __restrict__ W2,
                    const float*  __restrict__ bias,
                    float*        __restrict__ c,        // [B,H] in/out
                    __bf16*       __restrict__ h_out,    // [B,H] bf16
                    float*        __restrict__ h_out_f32)// [B,H] or nullptr
{
    __shared__ __bf16 ldsA[64 * (KC_ + 8)];    // 64 rows x padded K chunk (33 KB)

    const int lane  = threadIdx.x & 31;
    const int wave  = threadIdx.x >> 5;
    const int tileN = blockIdx.x * 8 + wave;    // 0..63  (hidden-unit tile)
    const int mB64  = blockIdx.y * 64;          // batch row block (64 rows)
    const int mRow  = lane & 15;
    const int khalf = lane >> 4;                // A/B lane half (K grouping)
    const int nCol  = tileN * 16 + (lane & 15); // global hidden unit (B rows / D col)

    v8f acc[4][MT_] = {};                       // [gate][mtile] fp32 accumulators

    const __bf16* Aptr[2] = {A1, A2};
    const __bf16* Wptr[2] = {W1, W2};
    const int     Ks[2]   = {K1, K2};

    for (int ph = 0; ph < 2; ++ph) {
        const __bf16* A = Aptr[ph];
        const __bf16* W = Wptr[ph];
        const int K = Ks[ph];
        for (int kc = 0; kc < K; kc += KC_) {
            const int CH = (K - kc) < KC_ ? (K - kc) : KC_;   // 32 or 256
            const int stride = CH + 8;                        // bf16 elems, 16B padded
            // ---- cooperative stage: A[mB64 .. mB64+63][kc .. kc+CH-1] -> LDS ----
            __syncthreads();
            {
                const int ch16  = CH >> 3;        // 16B chunks per row
                const int total = 64 * ch16;
                for (int i = threadIdx.x; i < total; i += 256) {
                    const int r = i / ch16, cu = i % ch16;
                    *(v4u*)&ldsA[r * stride + cu * 8] =
                        ldg_u4(A + (size_t)(mB64 + r) * K + kc + cu * 8);
                }
            }
            __syncthreads();
            // ---- compute over this K chunk ----
            for (int kk = 0; kk < CH; kk += 32) {
                // A fragments from LDS: row = mt*16+mRow, K = khalf*8 + {0..7,16..23}
                AFrag a[MT_];
#pragma unroll
                for (int mt = 0; mt < MT_; ++mt) {
                    const __bf16* lp = &ldsA[(mt * 16 + mRow) * stride + kk + khalf * 8];
                    a[mt].u[0] = *(const v4u*)(lp);
                    a[mt].u[1] = *(const v4u*)(lp + 16);
                }
#pragma unroll
                for (int gi = 0; gi < 4; ++gi) {
                    // B fragment: lane holds column nCol, K = khalf*8 + {0..7,16..23}
                    AFrag b;
                    const __bf16* wr =
                        W + (size_t)(gi * H_ + nCol) * K + kc + kk + khalf * 8;
                    b.u[0] = ldg_u4(wr);
                    b.u[1] = ldg_u4(wr + 16);
#pragma unroll
                    for (int mt = 0; mt < MT_; ++mt) {
                        acc[gi][mt] = __builtin_amdgcn_wmma_f32_16x16x32_bf16(
                            false, a[mt].v, false, b.v, (short)0, acc[gi][mt],
                            false, false);
                    }
                }
            }
        }
    }

    // Epilogue: LSTM cell. D layout: M = mt*16 + (lane/16)*8 + r, N = nCol.
    const int n  = nCol;
    const float bi = ldg_f(bias + n);
    const float bf = ldg_f(bias + H_ + n);
    const float bg = ldg_f(bias + 2 * H_ + n);
    const float bo = ldg_f(bias + 3 * H_ + n);
#pragma unroll
    for (int mt = 0; mt < MT_; ++mt) {
        const int mBase = mB64 + mt * 16 + khalf * 8;
#pragma unroll
        for (int r = 0; r < 8; ++r) {
            const int m = mBase + r;
            const float si = sigmoid_fast(acc[0][mt][r] + bi);
            const float sf = sigmoid_fast(acc[1][mt][r] + bf);
            const float tg = tanh_fast   (acc[2][mt][r] + bg);
            const float so = sigmoid_fast(acc[3][mt][r] + bo);
            const size_t idx = (size_t)m * H_ + n;
            const float c2 = sf * ldg_f(c + idx) + si * tg;
            const float h2 = so * tanh_fast(c2);
            stg_f(c + idx, c2);
            stg_bf(h_out + idx, (__bf16)h2);
            if (h_out_f32) stg_f(h_out_f32 + idx, h2);
        }
    }
}

// out[t] = h1 @ Wp^T + bp   (B x E, fp32, tiny: 8.4 M MACs)
__global__ __launch_bounds__(256)
void proj_kernel(const float* __restrict__ h, const float* __restrict__ Wp,
                 const float* __restrict__ bp, float* __restrict__ out)
{
    const int idx = blockIdx.x * 256 + threadIdx.x;   // B*E = 8192
    const int m = idx >> 5;
    const int e = idx & 31;
    const float4* hv = (const float4*)(h  + (size_t)m * H_);
    const float4* wv = (const float4*)(Wp + (size_t)e * H_);
    float s = 0.f;
    for (int k = 0; k < H_ / 4; ++k) {
        float4 a = hv[k], b = wv[k];
        s += a.x * b.x + a.y * b.y + a.z * b.z + a.w * b.w;
    }
    out[idx] = s + bp[e];
}

// h = tanh(z0 @ Wz^T + bz), write bf16 (one-time, fp32 math)
__global__ __launch_bounds__(256)
void init_h_kernel(const float* __restrict__ z0, const float* __restrict__ Wz,
                   const float* __restrict__ bz, __bf16* __restrict__ hbf)
{
    const int idx = blockIdx.x * 256 + threadIdx.x;   // B*H = 262144
    const int m = idx >> 10;
    const int n = idx & 1023;
    const float4* zv = (const float4*)(z0 + (size_t)m * Z_);
    const float4* wv = (const float4*)(Wz + (size_t)n * Z_);
    float s = bz[n];
    for (int k = 0; k < Z_ / 4; ++k) {
        float4 a = zv[k], b = wv[k];
        s += a.x * b.x + a.y * b.y + a.z * b.z + a.w * b.w;
    }
    hbf[idx] = (__bf16)tanhf(s);
}

__global__ void f32_to_bf16_kernel(const float* __restrict__ src,
                                   __bf16* __restrict__ dst, int n)
{
    const int i = blockIdx.x * 256 + threadIdx.x;
    if (i < n) dst[i] = (__bf16)src[i];
}

__global__ void zero_f32_kernel(float* __restrict__ p, int n)
{
    const int i = blockIdx.x * 256 + threadIdx.x;
    if (i < n) p[i] = 0.0f;
}

// Teacher-forced inputs: step 0 zeros, step t uses targets[t-1]; to bf16.
__global__ void build_x_kernel(const float* __restrict__ targets,
                               __bf16* __restrict__ xbf)
{
    const int i = blockIdx.x * 256 + threadIdx.x;    // T*B*E = 524288
    if (i < B_ * E_) xbf[i] = (__bf16)0.0f;
    else             xbf[i] = (__bf16)targets[i - B_ * E_];
}

// bias[l][n] = b_ih[l][n] + b_hh[l][n]
__global__ void bias_kernel(const float* __restrict__ bi0, const float* __restrict__ bh0,
                            const float* __restrict__ bi1, const float* __restrict__ bh1,
                            float* __restrict__ bias0, float* __restrict__ bias1)
{
    const int i = blockIdx.x * 256 + threadIdx.x;    // 4096
    if (i < G_) {
        bias0[i] = bi0[i] + bh0[i];
        bias1[i] = bi1[i] + bh1[i];
    }
}

extern "C" void kernel_launch(void* const* d_in, const int* in_sizes, int n_in,
                              void* d_out, int out_size, void* d_ws, size_t ws_size,
                              hipStream_t stream) {
    (void)in_sizes; (void)n_in; (void)out_size; (void)ws_size;
    // setup_inputs order:
    const float* z       = (const float*)d_in[0];   // (1,B,Z)
    const float* targets = (const float*)d_in[1];   // (T,B,E)
    /* d_in[2] = length (int), hardcoded T_ */
    const float* W_ih0 = (const float*)d_in[3];
    const float* W_hh0 = (const float*)d_in[4];
    const float* b_ih0 = (const float*)d_in[5];
    const float* b_hh0 = (const float*)d_in[6];
    const float* W_ih1 = (const float*)d_in[7];
    const float* W_hh1 = (const float*)d_in[8];
    const float* b_ih1 = (const float*)d_in[9];
    const float* b_hh1 = (const float*)d_in[10];
    const float* Wz0   = (const float*)d_in[11];
    const float* bz0   = (const float*)d_in[12];
    const float* Wz1   = (const float*)d_in[13];
    const float* bz1   = (const float*)d_in[14];
    const float* Wp    = (const float*)d_in[15];
    const float* bp    = (const float*)d_in[16];
    float* out = (float*)d_out;                     // (T,B,E) fp32

    // ---- workspace carve-up (256B aligned regions) ----
    char* w = (char*)d_ws;
    auto alloc = [&](size_t bytes) -> char* {
        char* p = w;
        w += (bytes + 255) & ~(size_t)255;
        return p;
    };
    __bf16* Wih0_bf = (__bf16*)alloc((size_t)G_ * E_ * 2);
    __bf16* Whh0_bf = (__bf16*)alloc((size_t)G_ * H_ * 2);
    __bf16* Wih1_bf = (__bf16*)alloc((size_t)G_ * H_ * 2);
    __bf16* Whh1_bf = (__bf16*)alloc((size_t)G_ * H_ * 2);
    __bf16* Xbf     = (__bf16*)alloc((size_t)T_ * B_ * E_ * 2);
    float*  bias0   = (float*) alloc((size_t)G_ * 4);
    float*  bias1   = (float*) alloc((size_t)G_ * 4);
    __bf16* h0bf[2] = { (__bf16*)alloc((size_t)B_ * H_ * 2),
                        (__bf16*)alloc((size_t)B_ * H_ * 2) };
    __bf16* h1bf[2] = { (__bf16*)alloc((size_t)B_ * H_ * 2),
                        (__bf16*)alloc((size_t)B_ * H_ * 2) };
    float*  h1f32   = (float*) alloc((size_t)B_ * H_ * 4);
    float*  c0      = (float*) alloc((size_t)B_ * H_ * 4);
    float*  c1      = (float*) alloc((size_t)B_ * H_ * 4);

    // ---- one-time prep (per launch; deterministic) ----
    auto cvt = [&](const float* src, __bf16* dst, int n) {
        f32_to_bf16_kernel<<<(n + 255) / 256, 256, 0, stream>>>(src, dst, n);
    };
    cvt(W_ih0, Wih0_bf, G_ * E_);
    cvt(W_hh0, Whh0_bf, G_ * H_);
    cvt(W_ih1, Wih1_bf, G_ * H_);
    cvt(W_hh1, Whh1_bf, G_ * H_);
    build_x_kernel<<<(T_ * B_ * E_ + 255) / 256, 256, 0, stream>>>(targets, Xbf);
    bias_kernel<<<(G_ + 255) / 256, 256, 0, stream>>>(b_ih0, b_hh0, b_ih1, b_hh1,
                                                      bias0, bias1);
    zero_f32_kernel<<<(B_ * H_ + 255) / 256, 256, 0, stream>>>(c0, B_ * H_);
    zero_f32_kernel<<<(B_ * H_ + 255) / 256, 256, 0, stream>>>(c1, B_ * H_);
    init_h_kernel<<<(B_ * H_) / 256, 256, 0, stream>>>(z, Wz0, bz0, h0bf[0]);
    init_h_kernel<<<(B_ * H_) / 256, 256, 0, stream>>>(z, Wz1, bz1, h1bf[0]);

    // ---- sequential recurrence ----
    const dim3 gridG(8, 4, 1);   // 32 WGs x 8 waves; wave = 64M x 16N x 4 gates
    for (int t = 0; t < T_; ++t) {
        const int cur = t & 1, nxt = (t + 1) & 1;
        // layer 0: gates = x_t @ Wih0^T + h0 @ Whh0^T + bias0
        lstm_gate_wmma<<<gridG, 256, 0, stream>>>(
            Xbf + (size_t)t * B_ * E_, E_, Wih0_bf,
            h0bf[cur], H_, Whh0_bf,
            bias0, c0, h0bf[nxt], (float*)nullptr);
        // layer 1: gates = h0_new @ Wih1^T + h1 @ Whh1^T + bias1
        lstm_gate_wmma<<<gridG, 256, 0, stream>>>(
            h0bf[nxt], H_, Wih1_bf,
            h1bf[cur], H_, Whh1_bf,
            bias1, c1, h1bf[nxt], h1f32);
        // out[t] = h1 @ Wp^T + bp  (fp32)
        proj_kernel<<<(B_ * E_) / 256, 256, 0, stream>>>(
            h1f32, Wp, bp, out + (size_t)t * B_ * E_);
    }
}
